// Seq2Seq_20572893348323
// MI455X (gfx1250) — matrix operand order repaired and checked by reading, not compiled, not run on hardware
//
#include <hip/hip_runtime.h>
#include <hip/hip_bf16.h>

typedef __attribute__((ext_vector_type(16))) _Float16 v16h;
typedef __attribute__((ext_vector_type(8)))  _Float16 v8h;
typedef __attribute__((ext_vector_type(8)))  float    v8f;
typedef _Float16 f16;

#define B_   128
#define TL_  400
#define AL_  50
#define QL_  32
#define E_   300
#define H_   300
#define D_   300
#define V_   30000
#define H2_  600
#define H3_  900

// Padded K strides (multiples of 32 f16 => 64B rows, b128-friendly)
#define KP300 320
#define KP600 608
#define KP900 928
#define KP1200 1216

#define SHUF16(a, b) __builtin_shufflevector(a, b, 0,1,2,3,4,5,6,7,8,9,10,11,12,13,14,15)

// ---------------------------------------------------------------------------
// 64x16 WMMA macro-tile (4 M-tiles per wave):  C = A[M,K] @ B[K,N] (+bias)
// A: row-major f16, row stride lda (>= K rounded up to 32, zero pad)
// Bt: B stored TRANSPOSED as [N, ldbt] f16, zero-padded for k in [K, ldbt)
// B fragment loaded once per K-step, reused by 4 WMMAs; fragments are
// double-buffered so loads for step k+1 overlap the 4 WMMAs of step k.
// ---------------------------------------------------------------------------
__device__ __forceinline__ void wmma_tile4(
    const f16* __restrict__ A, int lda,
    const f16* __restrict__ Bt, int ldbt,
    const float* __restrict__ bias,
    float* __restrict__ C, int ldc,
    int M, int N, int K, int tm4, int tn,
    int remap_b, int remap_ql)
{
  const int lane = threadIdx.x & 31;
  const int half = lane >> 4;     // 0: lanes 0-15, 1: lanes 16-31
  const int l16  = lane & 15;
  const int bcol = tn * 16 + l16;
  const int bcol_c = bcol < N ? bcol : N - 1;   // clamp: OOB cols never stored
  const f16* bp = Bt + (size_t)bcol_c * ldbt + half * 8;

  const f16* ap[4];
  #pragma unroll
  for (int s = 0; s < 4; ++s) {
    int arow = (tm4 * 4 + s) * 16 + l16;
    int arow_c = arow < M ? arow : M - 1;       // clamp: OOB rows never stored
    ap[s] = A + (size_t)arow_c * lda + half * 8;
  }

  v8f acc[4] = {};
  // prologue: load fragments for k0 = 0
  v8h cb0 = *(const v8h*)(bp);
  v8h cb1 = *(const v8h*)(bp + 16);
  v8h ca0[4], ca1[4];
  #pragma unroll
  for (int s = 0; s < 4; ++s) {
    ca0[s] = *(const v8h*)(ap[s]);
    ca1[s] = *(const v8h*)(ap[s] + 16);
  }

  for (int k0 = 0; k0 < K; k0 += 32) {
    // prefetch next K-step into distinct regs (uniform clamp; dummy on last)
    const int kn = k0 + 32;
    const int kl = (kn < K) ? kn : 0;
    v8h nb0 = *(const v8h*)(bp + kl);
    v8h nb1 = *(const v8h*)(bp + kl + 16);
    v8h na0[4], na1[4];
    #pragma unroll
    for (int s = 0; s < 4; ++s) {
      na0[s] = *(const v8h*)(ap[s] + kl);
      na1[s] = *(const v8h*)(ap[s] + kl + 16);
    }
    // consume current fragments: 4 back-to-back WMMAs sharing one B fragment
    v16h bv = SHUF16(cb0, cb1);
    #pragma unroll
    for (int s = 0; s < 4; ++s) {
      v16h av = SHUF16(ca0[s], ca1[s]);
      acc[s] = __builtin_amdgcn_wmma_f32_16x16x32_f16(
          false, av, false, bv, (short)0, acc[s], false, false);
    }
    cb0 = nb0; cb1 = nb1;
    #pragma unroll
    for (int s = 0; s < 4; ++s) { ca0[s] = na0[s]; ca1[s] = na1[s]; }
  }

  if (bcol < N) {
    const float bb = bias ? bias[bcol] : 0.f;
    #pragma unroll
    for (int s = 0; s < 4; ++s) {
      #pragma unroll
      for (int e = 0; e < 8; ++e) {
        int row = (tm4 * 4 + s) * 16 + half * 8 + e;  // C: VGPR e -> M = half*8+e
        if (row < M) {
          size_t off;
          if (remap_b > 0) {
            int b = row % remap_b, t = row / remap_b;
            off = ((size_t)b * remap_ql + t + 1) * (size_t)ldc + bcol;
          } else {
            off = (size_t)row * (size_t)ldc + bcol;
          }
          C[off] = acc[s][e] + bb;
        }
      }
    }
  }
}

// 4 waves / block, flattened macro-tile index
__global__ void gemm_f16_kernel(const f16* A, int lda, const f16* Bt, int ldbt,
                                const float* bias, float* C, int ldc,
                                int M, int N, int K, int remap_b, int remap_ql)
{
  const int ntn  = (N + 15) >> 4;
  const int ntm4 = (M + 63) >> 6;
  const int tile = blockIdx.x * (blockDim.x >> 5) + (threadIdx.x >> 5);
  if (tile >= ntn * ntm4) return;
  wmma_tile4(A, lda, Bt, ldbt, bias, C, ldc, M, N, K, tile / ntn, tile % ntn,
             remap_b, remap_ql);
}

// ---------------------------------------------------------------------------
// Encoder step: gridDim.z in {0:gi_fwd, 1:gh_fwd, 2:gi_bwd, 3:gh_bwd}
// gi = x_t @ Wih + bih ; gh = h @ Whh + bhh.  M=128, N=900, K=300.
// grid (57, 2, 4): 57 N-tiles x 2 M-macro-tiles x 4 slices, 1 wave/block.
// ---------------------------------------------------------------------------
struct EncArgs {
  const f16* xemb;                 // [B, T, KP300] f16
  const f16 *h16_f, *h16_b;        // [B, KP300] f16
  const f16 *Wih_f, *Whh_f, *Wih_b, *Whh_b;   // transposed [900, KP300] f16
  const float *bih_f, *bhh_f, *bih_b, *bhh_b; // [900] f32
  float *gi_f, *gh_f, *gi_b, *gh_b;           // [128,900] f32
  int t_f, t_b, T;
};

__global__ void enc_step_kernel(EncArgs a)
{
  const int z = blockIdx.z;
  const f16* A; int lda; const f16* Bt; const float* bias; float* C;
  if (z == 0)      { A = a.xemb + (size_t)a.t_f * KP300; lda = a.T * KP300; Bt = a.Wih_f; bias = a.bih_f; C = a.gi_f; }
  else if (z == 1) { A = a.h16_f;                        lda = KP300;       Bt = a.Whh_f; bias = a.bhh_f; C = a.gh_f; }
  else if (z == 2) { A = a.xemb + (size_t)a.t_b * KP300; lda = a.T * KP300; Bt = a.Wih_b; bias = a.bih_b; C = a.gi_b; }
  else             { A = a.h16_b;                        lda = KP300;       Bt = a.Whh_b; bias = a.bhh_b; C = a.gh_b; }
  wmma_tile4(A, lda, Bt, KP300, bias, C, H3_, B_, H3_, H_, blockIdx.y, blockIdx.x, 0, 0);
}

// GRU gate math for both directions; writes h (f32+f16) and rep slice (f16)
__global__ void enc_gate_kernel(const float* gi_f, const float* gh_f,
                                const float* gi_b, const float* gh_b,
                                float* h32_f, float* h32_b,
                                f16* h16_f, f16* h16_b,
                                f16* rep_h, int t_f, int t_b, int T)
{
  const int i = blockIdx.x * blockDim.x + threadIdx.x;
  if (i >= B_ * H_) return;
  const int dir = blockIdx.z;
  const float* gi = dir ? gi_b : gi_f;
  const float* gh = dir ? gh_b : gh_f;
  float* h32 = dir ? h32_b : h32_f;
  f16*   h16 = dir ? h16_b : h16_f;
  const int t = dir ? t_b : t_f;
  const int b = i / H_, j = i % H_;
  const size_t g = (size_t)b * H3_ + j;
  const float r  = 1.f / (1.f + expf(-(gi[g]         + gh[g])));
  const float zz = 1.f / (1.f + expf(-(gi[g + H_]    + gh[g + H_])));
  const float n  = tanhf(gi[g + 2 * H_] + r * gh[g + 2 * H_]);
  const float h  = (1.f - zz) * n + zz * h32[i];
  h32[i] = h;
  h16[(size_t)b * KP300 + j] = (f16)h;
  rep_h[((size_t)b * T + t) * KP600 + dir * H_ + j] = (f16)h;
}

// Decoder GRU gates; writes h (f32+f16) and h-part of out-cat buffer (f16)
__global__ void dec_gate_kernel(const float* gi, const float* gh,
                                float* h32, f16* h16, f16* outcat, int t)
{
  const int i = blockIdx.x * blockDim.x + threadIdx.x;
  if (i >= B_ * D_) return;
  const int b = i / D_, j = i % D_;
  const size_t g = (size_t)b * H3_ + j;
  const float r  = 1.f / (1.f + expf(-(gi[g]         + gh[g])));
  const float zz = 1.f / (1.f + expf(-(gi[g + D_]    + gh[g + D_])));
  const float n  = tanhf(gi[g + 2 * D_] + r * gh[g + 2 * D_]);
  const float h  = (1.f - zz) * n + zz * h32[i];
  h32[i] = h;
  h16[(size_t)b * KP300 + j] = (f16)h;
  outcat[((size_t)t * B_ + b) * KP1200 + j] = (f16)h;
}

// scores[b,s] = sum_d tanh(qproj[b,s,d] + hWk[b,d]) * v[d] ; one wave per (b,s)
__global__ void attn_scores_kernel(const float* qproj, const float* hWk,
                                   const float* v, float* scores)
{
  const int wid  = (blockIdx.x * blockDim.x + threadIdx.x) >> 5;
  const int lane = threadIdx.x & 31;
  if (wid >= B_ * TL_) return;
  const int b = wid / TL_, s = wid % TL_;
  const float* qp = qproj + ((size_t)b * TL_ + s) * D_;
  const float* hk = hWk + (size_t)b * D_;
  float sum = 0.f;
  for (int d = lane; d < D_; d += 32) sum += tanhf(qp[d] + hk[d]) * v[d];
  #pragma unroll
  for (int off = 16; off > 0; off >>= 1) sum += __shfl_xor(sum, off, 32);
  if (lane == 0) scores[(size_t)b * TL_ + s] = sum;
}

__global__ void softmax400_kernel(const float* scores, float* aw)
{
  const int b = blockIdx.x, tid = threadIdx.x;   // 256 threads
  __shared__ float red[256];
  const float* sc = scores + (size_t)b * TL_;
  float m = -1e30f;
  for (int s = tid; s < TL_; s += 256) m = fmaxf(m, sc[s]);
  red[tid] = m; __syncthreads();
  for (int o = 128; o > 0; o >>= 1) { if (tid < o) red[tid] = fmaxf(red[tid], red[tid + o]); __syncthreads(); }
  m = red[0]; __syncthreads();
  float sum = 0.f;
  for (int s = tid; s < TL_; s += 256) sum += expf(sc[s] - m);
  red[tid] = sum; __syncthreads();
  for (int o = 128; o > 0; o >>= 1) { if (tid < o) red[tid] += red[tid + o]; __syncthreads(); }
  const float inv = 1.f / red[0];
  for (int s = tid; s < TL_; s += 256) aw[(size_t)b * TL_ + s] = expf(sc[s] - m) * inv;
}

// weighted[b,e] = sum_s aw[b,s] * rep[b,s,e]; feeds rnn_in (f16) and out-cat (f16)
__global__ void attn_weighted_kernel(const float* aw, const f16* rep_h,
                                     f16* rnn_in, f16* outcat, int t)
{
  const int i = blockIdx.x * blockDim.x + threadIdx.x;
  if (i >= B_ * H2_) return;
  const int b = i / H2_, e = i % H2_;
  const float* a = aw + (size_t)b * TL_;
  const f16* r = rep_h + (size_t)b * TL_ * KP600 + e;
  float acc = 0.f;
  for (int s = 0; s < TL_; ++s) acc += a[s] * (float)r[(size_t)s * KP600];
  rnn_in[(size_t)b * KP900 + E_ + e]               = (f16)acc;
  outcat[((size_t)t * B_ + b) * KP1200 + E_ + e]   = (f16)acc;
}

// rnn_in[:, 0:E] = q_emb[:, t, :]
__global__ void dec_prep_emb_kernel(const f16* qemb, f16* rnn_in, int t)
{
  const int i = blockIdx.x * blockDim.x + threadIdx.x;
  if (i >= B_ * E_) return;
  const int b = i / E_, e = i % E_;
  rnn_in[(size_t)b * KP900 + e] = qemb[((size_t)b * QL_ + t) * E_ + e];
}

// outcat[t, b, 900:1200] = q_emb[b, t, :] for all decode steps
__global__ void outcat_emb_kernel(const f16* qemb, f16* outcat)
{
  const int i = blockIdx.x * blockDim.x + threadIdx.x;
  if (i >= (QL_ - 1) * B_ * E_) return;
  const int t = i / (B_ * E_), rem = i % (B_ * E_);
  const int b = rem / E_, e = rem % E_;
  outcat[((size_t)t * B_ + b) * KP1200 + 900 + e] = qemb[((size_t)b * QL_ + t) * E_ + e];
}

// Embedding gather into K-padded rows (pad zeroed)
__global__ void gather_pad_kernel(const int* tok, const float* emb, f16* out, int ntok)
{
  const size_t i = (size_t)blockIdx.x * blockDim.x + threadIdx.x;
  if (i >= (size_t)ntok * KP300) return;
  const size_t r = i / KP300, e = i % KP300;
  out[i] = (e < E_) ? (f16)emb[(size_t)tok[r] * E_ + e] : (f16)0.f;
}

// Plain gather (unpadded, for q_emb which is never a GEMM operand directly)
__global__ void gather_f16_kernel(const int* tok, const float* emb, f16* out, int ntok)
{
  const size_t i = (size_t)blockIdx.x * blockDim.x + threadIdx.x;
  if (i >= (size_t)ntok * E_) return;
  const size_t r = i / E_, e = i % E_;
  out[i] = (f16)emb[(size_t)tok[r] * E_ + e];
}

// Transpose-convert weight: src [K, N] f32  ->  dst [N, K32] f16, zero K-pad
__global__ void convT_kernel(const float* s, f16* d, int K, int N, int K32)
{
  const size_t i = (size_t)blockIdx.x * blockDim.x + threadIdx.x;
  if (i >= (size_t)N * K32) return;
  const int n = (int)(i / K32), k = (int)(i % K32);
  d[i] = (k < K) ? (f16)s[(size_t)k * N + n] : (f16)0.f;
}

// f32 [B,300] -> f16 [B,KP300] (pad untouched; pre-zeroed)
__global__ void conv_h_kernel(const float* s, f16* d)
{
  const int i = blockIdx.x * blockDim.x + threadIdx.x;
  if (i >= B_ * D_) return;
  const int b = i / D_, j = i % D_;
  d[(size_t)b * KP300 + j] = (f16)s[i];
}

__global__ void zero_f16_kernel(f16* p, size_t n)
{
  const size_t i = (size_t)blockIdx.x * blockDim.x + threadIdx.x;
  if (i < n) p[i] = (f16)0.f;
}

__global__ void zero_f32_kernel(float* p, size_t n)
{
  const size_t i = (size_t)blockIdx.x * blockDim.x + threadIdx.x;
  if (i < n) p[i] = 0.f;
}

// out[b, 0, :] = 0
__global__ void zero_t0_kernel(float* out)
{
  const size_t i = (size_t)blockIdx.x * blockDim.x + threadIdx.x;
  if (i >= (size_t)B_ * V_) return;
  const size_t b = i / V_, n = i % V_;
  out[(size_t)b * QL_ * V_ + n] = 0.f;
}

// ---------------------------------------------------------------------------
extern "C" void kernel_launch(void* const* d_in, const int* in_sizes, int n_in,
                              void* d_out, int out_size, void* d_ws, size_t ws_size,
                              hipStream_t stream)
{
  (void)in_sizes; (void)n_in; (void)out_size; (void)ws_size;

  const int*   text      = (const int*)d_in[0];
  const int*   answer    = (const int*)d_in[1];
  const int*   question  = (const int*)d_in[2];
  const float* embedding = (const float*)d_in[3];
  const float* te_Wih_f = (const float*)d_in[4],  *te_Whh_f = (const float*)d_in[5];
  const float* te_bih_f = (const float*)d_in[6],  *te_bhh_f = (const float*)d_in[7];
  const float* te_Wih_b = (const float*)d_in[8],  *te_Whh_b = (const float*)d_in[9];
  const float* te_bih_b = (const float*)d_in[10], *te_bhh_b = (const float*)d_in[11];
  const float* ae_Wih_f = (const float*)d_in[12], *ae_Whh_f = (const float*)d_in[13];
  const float* ae_bih_f = (const float*)d_in[14], *ae_bhh_f = (const float*)d_in[15];
  const float* ae_Wih_b = (const float*)d_in[16], *ae_Whh_b = (const float*)d_in[17];
  const float* ae_bih_b = (const float*)d_in[18], *ae_bhh_b = (const float*)d_in[19];
  const float* trans_W  = (const float*)d_in[20], *trans_b  = (const float*)d_in[21];
  const float* attn_Wk  = (const float*)d_in[22], *attn_Wq  = (const float*)d_in[23];
  const float* attn_b   = (const float*)d_in[24], *attn_v   = (const float*)d_in[25];
  const float* dec_Wih  = (const float*)d_in[26], *dec_Whh  = (const float*)d_in[27];
  const float* dec_bih  = (const float*)d_in[28], *dec_bhh  = (const float*)d_in[29];
  const float* out_W    = (const float*)d_in[30], *out_b    = (const float*)d_in[31];
  float* out = (float*)d_out;

  // ---- workspace carve-out ----
  char* wp = (char*)d_ws;
  auto alloc = [&](size_t elems, size_t esz) -> void* {
    void* r = (void*)wp;
    wp += (elems * esz + 255) & ~(size_t)255;
    return r;
  };
  // transposed + K-padded f16 weights [N, K32]
  f16* te_Wih_f_t = (f16*)alloc((size_t)H3_*KP300, 2);
  f16* te_Whh_f_t = (f16*)alloc((size_t)H3_*KP300, 2);
  f16* te_Wih_b_t = (f16*)alloc((size_t)H3_*KP300, 2);
  f16* te_Whh_b_t = (f16*)alloc((size_t)H3_*KP300, 2);
  f16* ae_Wih_f_t = (f16*)alloc((size_t)H3_*KP300, 2);
  f16* ae_Whh_f_t = (f16*)alloc((size_t)H3_*KP300, 2);
  f16* ae_Wih_b_t = (f16*)alloc((size_t)H3_*KP300, 2);
  f16* ae_Whh_b_t = (f16*)alloc((size_t)H3_*KP300, 2);
  f16* trans_W_t  = (f16*)alloc((size_t)D_*KP600, 2);
  f16* attn_Wk_t  = (f16*)alloc((size_t)D_*KP300, 2);
  f16* attn_Wq_t  = (f16*)alloc((size_t)D_*KP600, 2);
  f16* dec_Wih_t  = (f16*)alloc((size_t)H3_*KP900, 2);
  f16* dec_Whh_t  = (f16*)alloc((size_t)H3_*KP300, 2);
  f16* out_W_t    = (f16*)alloc((size_t)V_*KP1200, 2);
  // f16 activations (K-padded rows where used as GEMM A)
  f16* text_emb_h = (f16*)alloc((size_t)B_*TL_*KP300, 2);
  f16* ans_emb_h  = (f16*)alloc((size_t)B_*AL_*KP300, 2);
  f16* qemb_h     = (f16*)alloc((size_t)B_*QL_*E_, 2);
  f16* text_rep_h = (f16*)alloc((size_t)B_*TL_*KP600, 2);
  f16* ans_rep_h  = (f16*)alloc((size_t)B_*AL_*KP600, 2);
  f16* outcat_h   = (f16*)alloc((size_t)(QL_-1)*B_*KP1200, 2);
  f16* rnn_in_h   = (f16*)alloc((size_t)B_*KP900, 2);
  f16* te_h16_f = (f16*)alloc((size_t)B_*KP300, 2);
  f16* te_h16_b = (f16*)alloc((size_t)B_*KP300, 2);
  f16* ae_h16_f = (f16*)alloc((size_t)B_*KP300, 2);
  f16* ae_h16_b = (f16*)alloc((size_t)B_*KP300, 2);
  f16* dec_h16  = (f16*)alloc((size_t)B_*KP300, 2);
  // f32 buffers
  float* gi_f = (float*)alloc(B_*H3_, 4); float* gh_f = (float*)alloc(B_*H3_, 4);
  float* gi_b = (float*)alloc(B_*H3_, 4); float* gh_b = (float*)alloc(B_*H3_, 4);
  float* dgi  = (float*)alloc(B_*H3_, 4); float* dgh  = (float*)alloc(B_*H3_, 4);
  float* te_h32_f = (float*)alloc(B_*H_, 4); float* te_h32_b = (float*)alloc(B_*H_, 4);
  float* ae_h32_f = (float*)alloc(B_*H_, 4); float* ae_h32_b = (float*)alloc(B_*H_, 4);
  float* dec_h32  = (float*)alloc(B_*D_, 4);
  float* qproj    = (float*)alloc((size_t)B_*TL_*D_, 4);
  float* hWk      = (float*)alloc(B_*D_, 4);
  float* scores   = (float*)alloc((size_t)B_*TL_, 4);
  float* aw       = (float*)alloc((size_t)B_*TL_, 4);

  auto convT = [&](const float* s, f16* d, int K, int N, int K32) {
    size_t n = (size_t)N * K32;
    convT_kernel<<<(unsigned)((n + 255) / 256), 256, 0, stream>>>(s, d, K, N, K32);
  };
  auto zero16 = [&](f16* p, size_t n) {
    zero_f16_kernel<<<(unsigned)((n + 255) / 256), 256, 0, stream>>>(p, n);
  };
  auto zero32 = [&](float* p, size_t n) {
    zero_f32_kernel<<<(unsigned)((n + 255) / 256), 256, 0, stream>>>(p, n);
  };
  auto gemm = [&](const f16* A, int lda, const f16* Bt, int ldbt, const float* bias,
                  float* C, int ldc, int M, int N, int K, int rb, int rq) {
    int tiles = ((M + 63) / 64) * ((N + 15) / 16);
    gemm_f16_kernel<<<(tiles + 3) / 4, 128, 0, stream>>>(A, lda, Bt, ldbt, bias,
                                                         C, ldc, M, N, K, rb, rq);
  };

  // ---- setup: transpose-convert weights + embedding gathers ----
  convT(te_Wih_f, te_Wih_f_t, E_, H3_, KP300); convT(te_Whh_f, te_Whh_f_t, H_, H3_, KP300);
  convT(te_Wih_b, te_Wih_b_t, E_, H3_, KP300); convT(te_Whh_b, te_Whh_b_t, H_, H3_, KP300);
  convT(ae_Wih_f, ae_Wih_f_t, E_, H3_, KP300); convT(ae_Whh_f, ae_Whh_f_t, H_, H3_, KP300);
  convT(ae_Wih_b, ae_Wih_b_t, E_, H3_, KP300); convT(ae_Whh_b, ae_Whh_b_t, H_, H3_, KP300);
  convT(trans_W, trans_W_t, H2_, D_, KP600);
  convT(attn_Wk, attn_Wk_t, D_, D_, KP300);
  convT(attn_Wq, attn_Wq_t, H2_, D_, KP600);
  convT(dec_Wih, dec_Wih_t, H3_, H3_, KP900);
  convT(dec_Whh, dec_Whh_t, D_, H3_, KP300);
  convT(out_W, out_W_t, 1200, V_, KP1200);

  gather_pad_kernel<<<((size_t)B_*TL_*KP300 + 255) / 256, 256, 0, stream>>>(text, embedding, text_emb_h, B_*TL_);
  gather_pad_kernel<<<((size_t)B_*AL_*KP300 + 255) / 256, 256, 0, stream>>>(answer, embedding, ans_emb_h, B_*AL_);
  gather_f16_kernel<<<((size_t)B_*QL_*E_ + 255) / 256, 256, 0, stream>>>(question, embedding, qemb_h, B_*QL_);

  // zero padded staging buffers + initial hidden states (ws reused across replays)
  zero16(text_rep_h, (size_t)B_*TL_*KP600);
  zero16(ans_rep_h,  (size_t)B_*AL_*KP600);
  zero16(outcat_h,   (size_t)(QL_-1)*B_*KP1200);
  zero16(rnn_in_h,   (size_t)B_*KP900);
  zero16(te_h16_f, (size_t)B_*KP300); zero16(te_h16_b, (size_t)B_*KP300);
  zero16(ae_h16_f, (size_t)B_*KP300); zero16(ae_h16_b, (size_t)B_*KP300);
  zero16(dec_h16,  (size_t)B_*KP300);
  zero32(te_h32_f, B_*H_); zero32(te_h32_b, B_*H_);
  zero32(ae_h32_f, B_*H_); zero32(ae_h32_b, B_*H_);

  const dim3 enc_grid(57, 2, 4);          // ceil(900/16) x (128/64) x 4 slices
  const dim3 gate_grid(150, 1, 2);        // 150*256 = 38400 = B*H

  // ---- text encoder: 400 fused fwd/bwd steps ----
  for (int t = 0; t < TL_; ++t) {
    EncArgs ea;
    ea.xemb = text_emb_h;
    ea.h16_f = te_h16_f; ea.h16_b = te_h16_b;
    ea.Wih_f = te_Wih_f_t; ea.Whh_f = te_Whh_f_t;
    ea.Wih_b = te_Wih_b_t; ea.Whh_b = te_Whh_b_t;
    ea.bih_f = te_bih_f; ea.bhh_f = te_bhh_f;
    ea.bih_b = te_bih_b; ea.bhh_b = te_bhh_b;
    ea.gi_f = gi_f; ea.gh_f = gh_f; ea.gi_b = gi_b; ea.gh_b = gh_b;
    ea.t_f = t; ea.t_b = TL_ - 1 - t; ea.T = TL_;
    enc_step_kernel<<<enc_grid, 32, 0, stream>>>(ea);
    enc_gate_kernel<<<gate_grid, 256, 0, stream>>>(gi_f, gh_f, gi_b, gh_b,
        te_h32_f, te_h32_b, te_h16_f, te_h16_b, text_rep_h, t, TL_ - 1 - t, TL_);
  }

  // ---- answer encoder: 50 fused fwd/bwd steps ----
  for (int t = 0; t < AL_; ++t) {
    EncArgs ea;
    ea.xemb = ans_emb_h;
    ea.h16_f = ae_h16_f; ea.h16_b = ae_h16_b;
    ea.Wih_f = ae_Wih_f_t; ea.Whh_f = ae_Whh_f_t;
    ea.Wih_b = ae_Wih_b_t; ea.Whh_b = ae_Whh_b_t;
    ea.bih_f = ae_bih_f; ea.bhh_f = ae_bhh_f;
    ea.bih_b = ae_bih_b; ea.bhh_b = ae_bhh_b;
    ea.gi_f = gi_f; ea.gh_f = gh_f; ea.gi_b = gi_b; ea.gh_b = gh_b;
    ea.t_f = t; ea.t_b = AL_ - 1 - t; ea.T = AL_;
    enc_step_kernel<<<enc_grid, 32, 0, stream>>>(ea);
    enc_gate_kernel<<<gate_grid, 256, 0, stream>>>(gi_f, gh_f, gi_b, gh_b,
        ae_h32_f, ae_h32_b, ae_h16_f, ae_h16_b, ans_rep_h, t, AL_ - 1 - t, AL_);
  }

  // hidden0 = ans_rep[:, -1, :] @ trans_W + trans_b   -> dec_h32/dec_h16
  gemm(ans_rep_h + (size_t)(AL_ - 1) * KP600, AL_ * KP600, trans_W_t, KP600, trans_b,
       dec_h32, D_, B_, D_, H2_, 0, 0);
  conv_h_kernel<<<(B_*D_ + 255) / 256, 256, 0, stream>>>(dec_h32, dec_h16);

  // qproj = text_rep @ attn_Wq + attn_b   [51200, 300]
  gemm(text_rep_h, KP600, attn_Wq_t, KP600, attn_b, qproj, D_, B_*TL_, D_, H2_, 0, 0);

  // out-cat embedding slice for all decode steps
  outcat_emb_kernel<<<((QL_-1)*B_*E_ + 255) / 256, 256, 0, stream>>>(qemb_h, outcat_h);

  // ---- decoder: 31 steps ----
  for (int t = 0; t < QL_ - 1; ++t) {
    gemm(dec_h16, KP300, attn_Wk_t, KP300, nullptr, hWk, D_, B_, D_, D_, 0, 0);
    attn_scores_kernel<<<(B_*TL_ + 7) / 8, 256, 0, stream>>>(qproj, hWk, attn_v, scores);
    softmax400_kernel<<<B_, 256, 0, stream>>>(scores, aw);
    attn_weighted_kernel<<<(B_*H2_ + 255) / 256, 256, 0, stream>>>(aw, text_rep_h,
                                                                   rnn_in_h, outcat_h, t);
    dec_prep_emb_kernel<<<(B_*E_ + 255) / 256, 256, 0, stream>>>(qemb_h, rnn_in_h, t);
    gemm(rnn_in_h, KP900, dec_Wih_t, KP900, dec_bih, dgi, H3_, B_, H3_, H3_, 0, 0);
    gemm(dec_h16, KP300, dec_Whh_t, KP300, dec_bhh, dgh, H3_, B_, H3_, D_, 0, 0);
    dec_gate_kernel<<<(B_*D_ + 255) / 256, 256, 0, stream>>>(dgi, dgh, dec_h32,
                                                             dec_h16, outcat_h, t);
  }

  // ---- output projection, scattered straight into [B, QL, V] ----
  zero_t0_kernel<<<((size_t)B_*V_ + 255) / 256, 256, 0, stream>>>(out);
  gemm(outcat_h, KP1200, out_W_t, KP1200, out_b, out, V_,
       (QL_ - 1) * B_, V_, 1200, /*remap_b=*/B_, /*remap_ql=*/QL_);
}
